// OrganAwareSwitchViT_38852274159843
// MI455X (gfx1250) — compile-verified, matrix operand order checked by reading, not run on hardware
//
#include <hip/hip_runtime.h>
#include <hip/hip_bf16.h>

// ---------------------------------------------------------------------------
// OrganAwareSwitchViT forward — MI455X (gfx1250, wave32) optimized.
//
// The reference returns (logits, aux_org, probs, entropy). The Switch-MoE
// expert FFN output (out_flat) is DEAD w.r.t. these outputs, so ~98.6 GFLOP
// of expert GEMMs are skipped. Remaining work is HBM-bound: one 77 MB pass
// over the patch tokens (arith intensity ~4 FLOP/B -> ~3.3 us floor at
// 23.3 TB/s). The 768x8 router GEMM rides that stream via
// v_wmma_f32_16x16x32_bf16 with router weights as the (transposed, padded)
// A matrix, so each lane 0..15 ends up holding all 8 expert logits of one
// token in its accumulator — softmax/entropy need zero cross-lane traffic.
//
// B fragments are built DIRECTLY from global memory: the CDNA5 16-bit
// operand layout maps, per lane, to four contiguous float4 loads of one
// token row (lanes n and n+16 jointly consume the row's 128 B K-chunk), so
// no LDS round-trip / dscnt stall sits in front of the WMMA.
// ---------------------------------------------------------------------------

typedef __bf16 bf16_t;
typedef bf16_t v16bf __attribute__((ext_vector_type(16)));
typedef float  v8f   __attribute__((ext_vector_type(8)));

#define B_    128
#define T_    196
#define D_    768
#define E_    8
#define ORG_  5
#define NTOK  (B_ * T_)      /* 25088 */
#define KCH   (D_ / 32)      /* 24 K-chunks of 32 */

#define OFF_LOGITS 0         /* 128*100 = 12800  */
#define OFF_AUX    12800     /* 128*5   = 640    */
#define OFF_PROBS  13440     /* 25088*8 = 200704 */
#define OFF_ENT    214144    /* 25088            */

// CDNA5 16-bit A-matrix 16x32 per-lane K mapping (ISA 7.12.2):
// lanes 0-15 : regs 0-3 -> K 0..7,  regs 4-7 -> K 16..23
// lanes 16-31: regs 0-3 -> K 8..15, regs 4-7 -> K 24..31
__device__ __forceinline__ int kmap(int lane, int idx) {
  int j = idx >> 1, p = idx & 1;
  int base = (lane & 16) ? 8 : 0;
  return (j < 4) ? (base + 2 * j + p) : (16 + base + 2 * (j - 4) + p);
}

__global__ __launch_bounds__(256)
void router_softmax_kernel(const float* __restrict__ tokens,
                           const float* __restrict__ priors,
                           const float* __restrict__ rw,   // (773, 8)
                           const float* __restrict__ rb,   // (8,)
                           float* __restrict__ out) {
  // Pre-swizzled router-weight A fragments (constant for the whole kernel).
  __shared__ __align__(32) bf16_t sA[KCH][32][16];      // 24 KB

  const int tid = threadIdx.x;
  for (int i = tid; i < KCH * 32 * 16; i += 256) {
    int idx  = i & 15;
    int lane = (i >> 4) & 31;
    int kc   = i >> 9;
    int m    = lane & 15;                       // A row = expert (pad 8..15)
    int k    = kc * 32 + kmap(lane, idx);
    float v  = (m < E_) ? rw[k * E_ + m] : 0.0f;
    sA[kc][lane][idx] = (bf16_t)v;
  }
  __syncthreads();

  const int wave  = tid >> 5;
  const int lane  = tid & 31;
  const int tile0 = (blockIdx.x * 8 + wave) * 16;   // 196 blocks * 8 waves * 16 = 25088

  // Each lane owns token row n = lane&15 of the tile; the lane pair (n, n+16)
  // covers the row's 128 B K-chunk with the two halves of the B layout.
  const int n    = lane & 15;
  const int base = (lane & 16) ? 8 : 0;
  const int t_n  = tile0 + n;
  const int b_n  = t_n / T_;
  const float* row =
      tokens + ((long)b_n * (T_ + 1) + 1 + (t_n - b_n * T_)) * D_;  // skip CLS

  v8f acc = {0.f, 0.f, 0.f, 0.f, 0.f, 0.f, 0.f, 0.f};

  for (int kc = 0; kc < KCH; ++kc) {
    const float* p = row + kc * 32;
    // Four contiguous b128 loads: K = base..base+7 and 16+base..16+base+7.
    const float4 f0 = *reinterpret_cast<const float4*>(p + base);
    const float4 f1 = *reinterpret_cast<const float4*>(p + base + 4);
    const float4 f2 = *reinterpret_cast<const float4*>(p + 16 + base);
    const float4 f3 = *reinterpret_cast<const float4*>(p + 16 + base + 4);
    if (kc + 1 < KCH)  // next 128 B row chunk (one cacheline)
      __builtin_prefetch(row + (kc + 1) * 32, 0, 0);

    union { v16bf v; bf16_t e[16]; } bfrag;
#pragma unroll
    for (int j = 0; j < 4; ++j) {
      bfrag.e[0 + j] = (bf16_t)((const float*)&f0)[j];   // -> v_cvt_pk_bf16_f32
      bfrag.e[4 + j] = (bf16_t)((const float*)&f1)[j];
      bfrag.e[8 + j] = (bf16_t)((const float*)&f2)[j];
      bfrag.e[12 + j] = (bf16_t)((const float*)&f3)[j];
    }

    v16bf a = *reinterpret_cast<const v16bf*>(&sA[kc][lane][0]);

    // D(16x16) = rw^T(16x32) x tok(32x16) + acc   [EXEC all-ones here]
    acc = __builtin_amdgcn_wmma_f32_16x16x32_bf16(false, a, false, bfrag.v,
                                                  (short)0, acc, false, false);
  }

  // Lanes 0..15 each hold expert logits 0..7 of token (tile0+lane).
  if (lane < 16) {
    const int t = tile0 + lane;
    const int b = t / T_;
    float pr[ORG_];
#pragma unroll
    for (int o = 0; o < ORG_; ++o) pr[o] = priors[b * ORG_ + o];

    float lg[E_];
    float mx = -3.4e38f;
#pragma unroll
    for (int e = 0; e < E_; ++e) {
      float v = acc[e] + rb[e];
#pragma unroll
      for (int o = 0; o < ORG_; ++o) v += pr[o] * rw[(D_ + o) * E_ + e];
      lg[e] = v;
      mx = fmaxf(mx, v);
    }
    float s = 0.f;
#pragma unroll
    for (int e = 0; e < E_; ++e) { lg[e] = __expf(lg[e] - mx); s += lg[e]; }
    const float inv = 1.0f / s;
    float ent = 0.f;
    float p8[E_];
#pragma unroll
    for (int e = 0; e < E_; ++e) {
      float pe = lg[e] * inv;
      p8[e] = pe;
      ent -= pe * __logf(pe + 1e-12f);
    }
    float4* po = reinterpret_cast<float4*>(out + OFF_PROBS + (size_t)t * E_);
    po[0] = make_float4(p8[0], p8[1], p8[2], p8[3]);
    po[1] = make_float4(p8[4], p8[5], p8[6], p8[7]);
    out[OFF_ENT + t] = ent;
  }
}

// ---------------------------------------------------------------------------
// CLS head: LayerNorm(768) + (768x100) + (768x5). 128 rows; ~20 MFLOP total.
// ---------------------------------------------------------------------------
__global__ __launch_bounds__(256)
void cls_head_kernel(const float* __restrict__ tokens,
                     const float* __restrict__ ln_g, const float* __restrict__ ln_b,
                     const float* __restrict__ cls_w, const float* __restrict__ cls_b,
                     const float* __restrict__ aux_w, const float* __restrict__ aux_b,
                     float* __restrict__ out) {
  const int b   = blockIdx.x;
  const int tid = threadIdx.x;
  __shared__ float sX[D_];
  __shared__ float red[256];

  const float* row = tokens + (size_t)b * (T_ + 1) * D_;   // CLS token (t=0)

  float lsum = 0.f;
  for (int d = tid; d < D_; d += 256) { float v = row[d]; sX[d] = v; lsum += v; }
  red[tid] = lsum;
  __syncthreads();
  for (int s = 128; s > 0; s >>= 1) {
    if (tid < s) red[tid] += red[tid + s];
    __syncthreads();
  }
  const float mu = red[0] / (float)D_;
  __syncthreads();

  float lvar = 0.f;
  for (int d = tid; d < D_; d += 256) { float v = sX[d] - mu; lvar += v * v; }
  red[tid] = lvar;
  __syncthreads();
  for (int s = 128; s > 0; s >>= 1) {
    if (tid < s) red[tid] += red[tid + s];
    __syncthreads();
  }
  const float rstd = rsqrtf(red[0] / (float)D_ + 1e-5f);
  __syncthreads();

  for (int d = tid; d < D_; d += 256)
    sX[d] = (sX[d] - mu) * rstd * ln_g[d] + ln_b[d];
  __syncthreads();

  for (int c = tid; c < 105; c += 256) {
    if (c < 100) {
      float a = cls_b[c];
      for (int d = 0; d < D_; ++d) a += sX[d] * cls_w[d * 100 + c];
      out[OFF_LOGITS + b * 100 + c] = a;
    } else {
      int o = c - 100;
      float a = aux_b[o];
      for (int d = 0; d < D_; ++d) a += sX[d] * aux_w[d * ORG_ + o];
      out[OFF_AUX + b * ORG_ + o] = a;
    }
  }
}

extern "C" void kernel_launch(void* const* d_in, const int* in_sizes, int n_in,
                              void* d_out, int out_size, void* d_ws, size_t ws_size,
                              hipStream_t stream) {
  const float* tokens = (const float*)d_in[0];
  const float* priors = (const float*)d_in[1];
  const float* rw     = (const float*)d_in[2];
  const float* rb     = (const float*)d_in[3];
  // d_in[4..7] = w1,b1,w2,b2 : expert FFN is dead code w.r.t. returned outputs.
  const float* ln_g   = (const float*)d_in[8];
  const float* ln_b   = (const float*)d_in[9];
  const float* cls_w  = (const float*)d_in[10];
  const float* cls_b  = (const float*)d_in[11];
  const float* aux_w  = (const float*)d_in[12];
  const float* aux_b  = (const float*)d_in[13];
  float* out = (float*)d_out;

  // 25088 tokens / (8 waves * 16 tokens) = 196 blocks
  router_softmax_kernel<<<196, 256, 0, stream>>>(tokens, priors, rw, rb, out);
  cls_head_kernel<<<128, 256, 0, stream>>>(tokens, ln_g, ln_b,
                                           cls_w, cls_b, aux_w, aux_b, out);
}
